// RNNCellModel_17772574671608
// MI455X (gfx1250) — matrix operand order repaired and checked by reading, not compiled, not run on hardware
//
#include <hip/hip_runtime.h>
#include <hip/hip_bf16.h>

// ---------------------------------------------------------------------------
// LSTM LM for MI455X (gfx1250, wave32, WMMA).
//  - bf16 V_WMMA_F32_16X16X32_BF16 everywhere (8192 MACs/instr vs 1024 for
//    the f32 16x16x4 path); f32 accumulate + f32 gate math.
//  - Recurrent step: one fused kernel per t (GEMM [32x1536]x[1536x4096] with
//    K split across 4 waves, LDS reduction, sigmoid/tanh, c/h update).
//  - Decoders: one large WMMA GEMM over hs (16384x1024) x Wd^T (1024x2560).
//  - All weights (12.6+5.2 MB bf16) resident in the 192 MB L2 across the 512
//    sequential steps; 168 MB of decoder output is the only big HBM traffic.
// ---------------------------------------------------------------------------

typedef __bf16 bf16_t;
typedef __attribute__((ext_vector_type(16))) __bf16 v16bf;
typedef __attribute__((ext_vector_type(8)))  __bf16 v8bf;
typedef __attribute__((ext_vector_type(8)))  float  v8f;

#define NTOK0  2048
#define NTOK1  512
#define EMSIZE 256
#define NHID   1024
#define BB     32
#define TT     512
#define KCAT   1536                 // 2*EMSIZE + NHID
#define GDIM   4096                 // 4*NHID
#define NDEC   2560                 // NTOK0 + NTOK1
#define XROW   (2*EMSIZE)           // 512

union V16 { v16bf v; v8bf h[2]; __device__ V16() {} };

__device__ __forceinline__ v8bf load8(const bf16_t* p) {
    return *reinterpret_cast<const v8bf*>(p);
}
__device__ __forceinline__ float sigf(float x) {
    return 1.0f / (1.0f + __expf(-x));
}

// --------------------------- weight prep kernels ---------------------------

// Wcat[n][k]: k<512 -> W_ih[n][k], else W_hh[n][k-512]. bcat[n]=b_ih+b_hh.
__global__ void prep_wcat_kernel(const float* __restrict__ W_ih,
                                 const float* __restrict__ W_hh,
                                 const float* __restrict__ b_ih,
                                 const float* __restrict__ b_hh,
                                 bf16_t* __restrict__ Wcat,
                                 float* __restrict__ bcat) {
    size_t idx = (size_t)blockIdx.x * blockDim.x + threadIdx.x;
    const size_t total = (size_t)GDIM * KCAT;
    if (idx < total) {
        int n = (int)(idx / KCAT);
        int k = (int)(idx % KCAT);
        const float* src = (k < XROW) ? (W_ih + (size_t)n * XROW + k)
                                      : (W_hh + (size_t)n * NHID + (k - XROW));
        Wcat[idx] = (bf16_t)(*src);
    }
    if (idx < GDIM) bcat[idx] = b_ih[idx] + b_hh[idx];
}

// Wd[n][k]: n<2048 -> Wd0[n][k] else Wd1[n-2048][k]
__global__ void prep_wd_kernel(const float* __restrict__ Wd0,
                               const float* __restrict__ Wd1,
                               bf16_t* __restrict__ Wd) {
    size_t idx = (size_t)blockIdx.x * blockDim.x + threadIdx.x;
    const size_t total = (size_t)NDEC * NHID;
    if (idx < total) {
        int n = (int)(idx / NHID);
        int k = (int)(idx % NHID);
        const float* src = (n < NTOK0) ? (Wd0 + (size_t)n * NHID + k)
                                       : (Wd1 + (size_t)(n - NTOK0) * NHID + k);
        Wd[idx] = (bf16_t)(*src);
    }
}

// X[t][b][0:256]=emb0[tok0[b][t]], X[t][b][256:512]=emb1[tok1[b][t]]   (bf16)
__global__ void embed_kernel(const int* __restrict__ tok0,
                             const int* __restrict__ tok1,
                             const float* __restrict__ emb0,
                             const float* __restrict__ emb1,
                             bf16_t* __restrict__ X) {
    int bt = blockIdx.x;            // 0..B*T-1  = b*T + t
    int b  = bt >> 9;               // /512
    int t  = bt & (TT - 1);
    int tid = threadIdx.x;          // 256 threads
    int i0 = tok0[(size_t)b * TT + t];
    int i1 = tok1[(size_t)b * TT + t];
    bf16_t* dst = X + ((size_t)t * BB + b) * XROW;
    dst[tid]          = (bf16_t)emb0[(size_t)i0 * EMSIZE + tid];
    dst[tid + EMSIZE] = (bf16_t)emb1[(size_t)i1 * EMSIZE + tid];
}

__global__ void init_state_kernel(const float* __restrict__ h0,
                                  const float* __restrict__ c0,
                                  bf16_t* __restrict__ hbf0,
                                  float* __restrict__ hf32,
                                  float* __restrict__ cf32) {
    int i = blockIdx.x * blockDim.x + threadIdx.x;   // 0..32767
    float h = h0[i], c = c0[i];
    hbf0[i] = (bf16_t)h;
    hf32[i] = h;
    cf32[i] = c;
}

// ------------------------------- LSTM step ---------------------------------
// grid = 64 blocks (one 16-col hidden block each), block = 128 threads (4
// waves). Each wave owns K-range [w*384, w*384+384) and accumulates 8 tiles:
// 4 gates x 2 M-tiles (rows 0..31). Partials reduced through LDS, then the
// block does bias + sigmoid/tanh + c/h update for its 32x16 hidden slice.
__global__ void lstm_step_kernel(const bf16_t* __restrict__ X,    // [T][B][512]
                                 const bf16_t* __restrict__ Wcat, // [4096][1536]
                                 const float*  __restrict__ bcat, // [4096]
                                 const bf16_t* __restrict__ h_rd, // [B][1024]
                                 bf16_t* __restrict__ h_wr,       // [B][1024]
                                 float*  __restrict__ h_f32,      // [B][1024]
                                 float*  __restrict__ c_f32,      // [B][1024]
                                 bf16_t* __restrict__ hs_bf,      // [T*B][1024]
                                 int t) {
    __shared__ float lds[4][4 * BB * 16];   // [wave][gate*32*16 + m*16 + n]

    const int tid  = threadIdx.x;
    const int wave = tid >> 5;              // 0..3  (K split)
    const int lane = tid & 31;
    const int cb   = blockIdx.x;            // 0..63 hidden column block

    const bf16_t* xrow = X + (size_t)t * BB * XROW;

    const int row_a  = lane & 15;
    const int koff_a = (lane >> 4) << 3;    // 0 or 8  (A frag K offset)
    const int koff_b = (lane >> 4) << 3;    // 0 or 8  (B frag K offset)

    // K base is uniform across the wave: force it scalar so the x/h boundary
    // test below is an SALU branch and EXEC stays all-1s around the WMMAs.
    const int kbase = __builtin_amdgcn_readfirstlane(wave * 384);

    v8f acc[4][2];
#pragma unroll
    for (int g = 0; g < 4; ++g)
#pragma unroll
        for (int m = 0; m < 2; ++m)
            acc[g][m] = v8f{0.f, 0.f, 0.f, 0.f, 0.f, 0.f, 0.f, 0.f};

#pragma unroll 2
    for (int kk = 0; kk < 12; ++kk) {
        const int k0 = kbase + kk * 32;          // scalar

        // Branchless A-pointer select (x_t for k0<512, h_{t-1} otherwise).
        const bf16_t* p0 = (k0 < XROW)
            ? (xrow + (size_t)row_a * XROW + k0 + koff_a)
            : (h_rd + (size_t)row_a * NHID + (k0 - XROW) + koff_a);
        const bf16_t* p1 = (k0 < XROW)
            ? (xrow + (size_t)(row_a + 16) * XROW + k0 + koff_a)
            : (h_rd + (size_t)(row_a + 16) * NHID + (k0 - XROW) + koff_a);

        V16 a0, a1;
        a0.h[0] = load8(p0);  a0.h[1] = load8(p0 + 16);
        a1.h[0] = load8(p1);  a1.h[1] = load8(p1 + 16);

#pragma unroll
        for (int g = 0; g < 4; ++g) {
            const int ncol = cb * 16 + g * NHID + (lane & 15); // Wcat row
            const bf16_t* bp = Wcat + (size_t)ncol * KCAT + k0 + koff_b;
            V16 bf;
            bf.h[0] = load8(bp);
            bf.h[1] = load8(bp + 16);
            acc[g][0] = __builtin_amdgcn_wmma_f32_16x16x32_bf16(
                false, a0.v, false, bf.v, (short)0, acc[g][0], false, false);
            acc[g][1] = __builtin_amdgcn_wmma_f32_16x16x32_bf16(
                false, a1.v, false, bf.v, (short)0, acc[g][1], false, false);
        }
    }

    // dump partials: C/D layout -> m = v + 8*(lane>>4), n = lane&15
    {
        float* my = &lds[wave][0];
        const int n = lane & 15;
        const int mhi = (lane >> 4) << 3;
#pragma unroll
        for (int g = 0; g < 4; ++g)
#pragma unroll
            for (int mt = 0; mt < 2; ++mt)
#pragma unroll
                for (int v = 0; v < 8; ++v) {
                    int m = mt * 16 + mhi + v;
                    my[(g * BB + m) * 16 + n] = acc[g][mt][v];
                }
    }
    __syncthreads();

    // elementwise: 512 (m,n) cells / 128 threads = 4 each
#pragma unroll
    for (int e = tid; e < BB * 16; e += 128) {
        const int m = e >> 4;               // batch row
        const int n = e & 15;
        const int colh = cb * 16 + n;       // hidden index
        float iv = 0.f, fv = 0.f, gv = 0.f, ov = 0.f;
#pragma unroll
        for (int w = 0; w < 4; ++w) {
            iv += lds[w][(0 * BB + m) * 16 + n];
            fv += lds[w][(1 * BB + m) * 16 + n];
            gv += lds[w][(2 * BB + m) * 16 + n];
            ov += lds[w][(3 * BB + m) * 16 + n];
        }
        iv = sigf(iv + bcat[colh]);
        fv = sigf(fv + bcat[colh + NHID]);
        gv = tanhf(gv + bcat[colh + 2 * NHID]);
        ov = sigf(ov + bcat[colh + 3 * NHID]);

        const size_t hidx = (size_t)m * NHID + colh;
        float cnew = fv * c_f32[hidx] + iv * gv;
        float hnew = ov * tanhf(cnew);
        c_f32[hidx] = cnew;
        h_f32[hidx] = hnew;
        h_wr[hidx]  = (bf16_t)hnew;
        hs_bf[((size_t)t * BB + m) * NHID + colh] = (bf16_t)hnew;
    }
}

// ------------------------------- decoder GEMM ------------------------------
// hs[16384,1024](bf16, [t][b] rows) x Wd^T -> dec0/dec1 (f32, [b][t][v]).
// One wave per (32-row M pair, 16-col N tile): 32 K-iters x 2 WMMA, B reused.
__global__ void decoder_kernel(const bf16_t* __restrict__ hs,   // [16384][1024]
                               const bf16_t* __restrict__ Wd,   // [2560][1024]
                               const float*  __restrict__ bd0,  // [2048]
                               const float*  __restrict__ bd1,  // [512]
                               float* __restrict__ out) {
    const int lane = threadIdx.x & 31;
    const int wib  = __builtin_amdgcn_readfirstlane(threadIdx.x >> 5);
    const int wid  = blockIdx.x * (blockDim.x >> 5) + wib;
    const int NT   = NDEC / 16;              // 160
    const int mp   = wid / NT;               // 0..511  rows mp*32..mp*32+31
    const int nt   = wid % NT;               // 0..159

    const int row_a  = lane & 15;
    const int koff_a = (lane >> 4) << 3;
    const int koff_b = (lane >> 4) << 3;
    const int mb = mp * 32;

    v8f acc0 = v8f{0.f, 0.f, 0.f, 0.f, 0.f, 0.f, 0.f, 0.f};
    v8f acc1 = acc0;

#pragma unroll 4
    for (int k0 = 0; k0 < NHID; k0 += 32) {
        const bf16_t* p0 = hs + (size_t)(mb + row_a) * NHID + k0 + koff_a;
        const bf16_t* p1 = hs + (size_t)(mb + 16 + row_a) * NHID + k0 + koff_a;
        V16 a0, a1, bf;
        a0.h[0] = load8(p0);  a0.h[1] = load8(p0 + 16);
        a1.h[0] = load8(p1);  a1.h[1] = load8(p1 + 16);

        const int ncol = nt * 16 + (lane & 15);
        const bf16_t* bp = Wd + (size_t)ncol * NHID + k0 + koff_b;
        bf.h[0] = load8(bp);
        bf.h[1] = load8(bp + 16);

        acc0 = __builtin_amdgcn_wmma_f32_16x16x32_bf16(
            false, a0.v, false, bf.v, (short)0, acc0, false, false);
        acc1 = __builtin_amdgcn_wmma_f32_16x16x32_bf16(
            false, a1.v, false, bf.v, (short)0, acc1, false, false);
    }

    // Bias via address-select (single load, no EXEC manipulation).
    const int col = nt * 16 + (lane & 15);
    const float* bptr = (col < NTOK0) ? (bd0 + col) : (bd1 + (col - NTOK0));
    const float bias = *bptr;
#pragma unroll
    for (int v = 0; v < 8; ++v) { acc0[v] += bias; acc1[v] += bias; }

    const size_t DEC1_OFF = (size_t)BB * TT * NTOK0;
    const int mhi = (lane >> 4) << 3;
#pragma unroll
    for (int mt = 0; mt < 2; ++mt) {
        const v8f& a = mt ? acc1 : acc0;
#pragma unroll
        for (int v = 0; v < 8; ++v) {
            const int r = mb + mt * 16 + mhi + v;  // hs row = t*32 + b
            const int tt = r >> 5;
            const int b  = r & 31;
            // Store via index-select (single store, no EXEC manipulation).
            const size_t oidx = (col < NTOK0)
                ? ((size_t)b * TT * NTOK0 + (size_t)tt * NTOK0 + col)
                : (DEC1_OFF + (size_t)b * TT * NTOK1 + (size_t)tt * NTOK1 +
                   (col - NTOK0));
            out[oidx] = a[v];
        }
    }
}

__global__ void final_state_kernel(const float* __restrict__ hf32,
                                   const float* __restrict__ cf32,
                                   float* __restrict__ out) {
    int i = blockIdx.x * blockDim.x + threadIdx.x;   // 0..32767
    const size_t OFF_H = (size_t)BB * TT * NTOK0 + (size_t)BB * TT * NTOK1;
    out[OFF_H + i]              = hf32[i];
    out[OFF_H + BB * NHID + i]  = cf32[i];
}

// ------------------------------- launcher ----------------------------------

extern "C" void kernel_launch(void* const* d_in, const int* in_sizes, int n_in,
                              void* d_out, int out_size, void* d_ws, size_t ws_size,
                              hipStream_t stream) {
    const int*   tok0 = (const int*)d_in[0];
    const int*   tok1 = (const int*)d_in[1];
    const float* h0   = (const float*)d_in[2];
    const float* c0   = (const float*)d_in[3];
    const float* emb0 = (const float*)d_in[4];
    const float* emb1 = (const float*)d_in[5];
    const float* W_ih = (const float*)d_in[6];
    const float* W_hh = (const float*)d_in[7];
    const float* b_ih = (const float*)d_in[8];
    const float* b_hh = (const float*)d_in[9];
    const float* Wd0  = (const float*)d_in[10];
    const float* bd0  = (const float*)d_in[11];
    const float* Wd1  = (const float*)d_in[12];
    const float* bd1  = (const float*)d_in[13];
    float* out = (float*)d_out;

    // workspace layout (all 256B aligned), total ~68.6 MB
    char* ws = (char*)d_ws;
    size_t off = 0;
    auto take = [&](size_t bytes) -> char* {
        char* p = ws + off;
        off = (off + bytes + 255) & ~(size_t)255;
        return p;
    };
    bf16_t* Wcat = (bf16_t*)take((size_t)GDIM * KCAT * 2);   // 12.58 MB
    bf16_t* Wd   = (bf16_t*)take((size_t)NDEC * NHID * 2);   //  5.24 MB
    float*  bcat = (float*) take((size_t)GDIM * 4);
    bf16_t* X    = (bf16_t*)take((size_t)TT * BB * XROW * 2);// 16.78 MB
    bf16_t* hs   = (bf16_t*)take((size_t)TT * BB * NHID * 2);// 33.55 MB
    bf16_t* hbf  = (bf16_t*)take((size_t)2 * BB * NHID * 2); // ping-pong
    float*  hf32 = (float*) take((size_t)BB * NHID * 4);
    float*  cf32 = (float*) take((size_t)BB * NHID * 4);

    // 1) weight / bias / embedding prep (parallel, independent)
    {
        const size_t total = (size_t)GDIM * KCAT;
        prep_wcat_kernel<<<(unsigned)((total + 255) / 256), 256, 0, stream>>>(
            W_ih, W_hh, b_ih, b_hh, Wcat, bcat);
    }
    {
        const size_t total = (size_t)NDEC * NHID;
        prep_wd_kernel<<<(unsigned)((total + 255) / 256), 256, 0, stream>>>(
            Wd0, Wd1, Wd);
    }
    embed_kernel<<<BB * TT, EMSIZE, 0, stream>>>(tok0, tok1, emb0, emb1, X);
    init_state_kernel<<<(BB * NHID) / 256, 256, 0, stream>>>(h0, c0, hbf, hf32, cf32);

    // 2) sequential LSTM steps (stream order carries the t->t+1 dependency)
    for (int t = 0; t < TT; ++t) {
        bf16_t* hrd = hbf + (size_t)(t & 1) * (BB * NHID);
        bf16_t* hwr = hbf + (size_t)((t + 1) & 1) * (BB * NHID);
        lstm_step_kernel<<<NHID / 16, 128, 0, stream>>>(
            X, Wcat, bcat, hrd, hwr, hf32, cf32, hs, t);
    }

    // 3) decoder GEMM: waves = (16384/32) * (2560/16) = 81920; 8 waves/block
    decoder_kernel<<<81920 / 8, 256, 0, stream>>>(hs, Wd, bd0, bd1, out);

    // 4) final h_t / c_t
    final_state_kernel<<<(BB * NHID) / 256, 256, 0, stream>>>(hf32, cf32, out);
}
